// memory_3985729651377
// MI455X (gfx1250) — matrix operand order repaired and checked by reading, not compiled, not run on hardware
//
#include <hip/hip_runtime.h>
#include <hip/hip_bf16.h>
#include <math.h>

typedef __attribute__((ext_vector_type(8)))  __bf16 v8bf;
typedef __attribute__((ext_vector_type(16))) __bf16 v16bf;
typedef __attribute__((ext_vector_type(8)))  float  v8f;
typedef __attribute__((ext_vector_type(4)))  int    v4i;

#define B_ 4
#define T_ 8192
#define D_ 1024
#define S_ 32
#define L_ 256

#if __has_builtin(__builtin_amdgcn_global_load_async_to_lds_b128)
#define HAVE_ASYNC_LDS 1
#else
#define HAVE_ASYNC_LDS 0
#endif

#define AS1 __attribute__((address_space(1)))
#define AS3 __attribute__((address_space(3)))

__device__ __forceinline__ void async_wait0() {
#if HAVE_ASYNC_LDS
#if __has_builtin(__builtin_amdgcn_s_wait_asynccnt)
    __builtin_amdgcn_s_wait_asynccnt(0);
#else
    asm volatile("s_wait_asynccnt 0" ::: "memory");
#endif
#endif
}

__device__ __forceinline__ unsigned short f2bf(float f) {
    unsigned u = __float_as_uint(f);
    unsigned r = ((u >> 16) & 1u) + 0x7FFFu;   // round-to-nearest-even
    return (unsigned short)((u + r) >> 16);
}
__device__ __forceinline__ float bf2f(unsigned short h) {
    return __uint_as_float(((unsigned)h) << 16);
}

// ---------------------------------------------------------------- conversion
__global__ void k_f32_to_bf16(const float* __restrict__ in,
                              unsigned short* __restrict__ out, size_t n) {
    size_t i = (size_t)blockIdx.x * blockDim.x + threadIdx.x;
    size_t stride = (size_t)gridDim.x * blockDim.x;
    for (; i < n; i += stride) out[i] = f2bf(in[i]);
}

// ------------------------------------------------- big bf16 WMMA GEMM (NT)
// C[M,N] = A[M,K] @ W[N,K]^T, bf16 in, fp32 accumulate.
// Double-buffered LDS staging via GLOBAL_LOAD_ASYNC_TO_LDS_B128 (ASYNCcnt)
// when available; synchronous b128 staging otherwise.
// EPI==1: out = bf16(sigmoid(C + bias[n])) -> outB ; EPI==0: fp32 C -> outF.
template <int EPI>
__global__ __launch_bounds__(256)
void k_gemm_bf16(const unsigned short* __restrict__ A,
                 const unsigned short* __restrict__ W,
                 const float* __restrict__ bias,
                 float* __restrict__ outF,
                 unsigned short* __restrict__ outB,
                 int M, int N, int K) {
    __shared__ unsigned short As[2][128][72];   // [buf][m][k], padded rows
    __shared__ unsigned short Bs[2][128][72];   // [buf][n][k]

    const int tid  = threadIdx.x;
    const int lane = tid & 31;
    const int wid  = tid >> 5;
    const int waveM = wid & 3;          // 4 waves along M (32 rows each)
    const int waveN = wid >> 2;         // 2 waves along N (64 cols each)
    const int l15 = lane & 15;
    const int hi  = lane >> 4;          // lane half select
    const int bm0 = blockIdx.y * 128;
    const int bn0 = blockIdx.x * 128;
    const int sr  = tid >> 1;           // staging row (0..127)
    const int sc  = (tid & 1) * 32;     // staging col chunk (halves)
    const int NT  = K >> 6;             // number of 64-wide k tiles

    v8f c[2][4];
#pragma unroll
    for (int i = 0; i < 2; i++)
#pragma unroll
        for (int j = 0; j < 4; j++)
#pragma unroll
            for (int e = 0; e < 8; e++) c[i][j][e] = 0.0f;

    // ---- stage one 128x64 A tile + 128x64 B tile into LDS buffer `buf`
    auto stage = [&](int buf, int kt) {
        const unsigned short* ag = A + (size_t)(bm0 + sr) * K + kt + sc;
        const unsigned short* bg = W + (size_t)(bn0 + sr) * K + kt + sc;
#if HAVE_ASYNC_LDS
#pragma unroll
        for (int i = 0; i < 4; i++) {
            __builtin_amdgcn_global_load_async_to_lds_b128(
                (AS1 v4i*)(ag + i * 8),
                (AS3 v4i*)&As[buf][sr][sc + i * 8], 0, 0);
            __builtin_amdgcn_global_load_async_to_lds_b128(
                (AS1 v4i*)(bg + i * 8),
                (AS3 v4i*)&Bs[buf][sr][sc + i * 8], 0, 0);
        }
#else
        uint4 av[4], bv[4];
#pragma unroll
        for (int i = 0; i < 4; i++) {
            av[i] = ((const uint4*)ag)[i];
            bv[i] = ((const uint4*)bg)[i];
        }
#pragma unroll
        for (int i = 0; i < 4; i++) {
            *(uint4*)&As[buf][sr][sc + i * 8] = av[i];
            *(uint4*)&Bs[buf][sr][sc + i * 8] = bv[i];
        }
#endif
    };

    stage(0, 0);                         // prologue: fill buffer 0

    for (int t = 0; t < NT; ++t) {
        const int cur = t & 1;
        async_wait0();                   // buffer `cur` fully landed in LDS
        __syncthreads();                 // visible to all waves (fences DS too)
        if (t + 1 < NT) stage(cur ^ 1, (t + 1) * 64);   // overlap next DMA

#pragma unroll
        for (int kk = 0; kk < 64; kk += 32) {
            v16bf afr[2], bfr[4];
#pragma unroll
            for (int fm = 0; fm < 2; fm++) {
                const int row = waveM * 32 + fm * 16 + l15;
                // A 16x32 bf16 layout: lane<16 -> K{0..7,16..23}; lane>=16 -> K{8..15,24..31}
                v8bf lo = *(const v8bf*)&As[cur][row][kk + hi * 8];
                v8bf hh = *(const v8bf*)&As[cur][row][kk + 16 + hi * 8];
                afr[fm] = __builtin_shufflevector(lo, hh, 0,1,2,3,4,5,6,7,8,9,10,11,12,13,14,15);
            }
#pragma unroll
            for (int fn = 0; fn < 4; fn++) {
                const int nn = waveN * 64 + fn * 16 + l15;
                // B 32x16 bf16 layout: lane<16 -> K{0..7,16..23}; lane>=16 -> K{8..15,24..31}
                v8bf lo = *(const v8bf*)&Bs[cur][nn][kk + hi * 8];
                v8bf hh = *(const v8bf*)&Bs[cur][nn][kk + 16 + hi * 8];
                bfr[fn] = __builtin_shufflevector(lo, hh, 0,1,2,3,4,5,6,7,8,9,10,11,12,13,14,15);
            }
#pragma unroll
            for (int fm = 0; fm < 2; fm++)
#pragma unroll
                for (int fn = 0; fn < 4; fn++)
                    c[fm][fn] = __builtin_amdgcn_wmma_f32_16x16x32_bf16(
                        false, afr[fm], false, bfr[fn], (short)0, c[fm][fn], false, false);
        }
        __syncthreads();                 // readers done before buf reuse
    }

#pragma unroll
    for (int fm = 0; fm < 2; fm++)
#pragma unroll
        for (int fn = 0; fn < 4; fn++) {
            const int colg = bn0 + waveN * 64 + fn * 16 + l15;
#pragma unroll
            for (int rr = 0; rr < 8; rr++) {
                const int rowg = bm0 + waveM * 32 + fm * 16 + hi * 8 + rr;
                float v = c[fm][fn][rr];
                if (EPI == 1) {
                    v += bias[colg];
                    v = 1.0f / (1.0f + expf(-v));
                    outB[(size_t)rowg * N + colg] = f2bf(v);
                } else {
                    outF[(size_t)rowg * N + colg] = v;
                }
            }
        }
}

// -------------------------------------------------- tiny fp32 GEMMs (NT/NN)
__global__ void k_sgemm_nt(const float* __restrict__ A, const float* __restrict__ Bw,
                           float* __restrict__ C, int M, int N, int K) {
    int idx = blockIdx.x * blockDim.x + threadIdx.x;
    if (idx >= M * N) return;
    int m = idx / N, n = idx % N;
    const float4* a = (const float4*)(A + (size_t)m * K);
    const float4* b = (const float4*)(Bw + (size_t)n * K);
    float acc = 0.f;
    for (int i = 0; i < K / 4; i++) {
        float4 x = a[i], y = b[i];
        acc += x.x * y.x + x.y * y.y + x.z * y.z + x.w * y.w;
    }
    C[idx] = acc;
}

__global__ void k_sgemm_nn(const float* __restrict__ A, const float* __restrict__ Bw,
                           float* __restrict__ C, int M, int N, int K) {
    int idx = blockIdx.x * blockDim.x + threadIdx.x;
    if (idx >= M * N) return;
    int m = idx / N, n = idx % N;
    float acc = 0.f;
    for (int k = 0; k < K; k++) acc += A[(size_t)m * K + k] * Bw[(size_t)k * N + n];
    C[idx] = acc;
}

// ------------------------------------- write attention: xbar = softmax(qk.x).x
__global__ __launch_bounds__(256)
void k_write_pass(const float* __restrict__ x, const float* __restrict__ qk,
                  float* __restrict__ xbar) {
    __shared__ float red[256];
    __shared__ float attn[256];
    const int bs = blockIdx.x;                 // b*32+s
    const int b = bs >> 5, s = bs & 31;
    const int l = threadIdx.x;                 // token within segment (L=256)
    const float scale = 0.03125f;              // 1/sqrt(1024)
    const float* xbase = x + ((size_t)b * T_ + (size_t)s * L_) * D_;

    const float4* xr = (const float4*)(xbase + (size_t)l * D_);
    const float4* qr = (const float4*)(qk + (size_t)s * D_);
    float acc = 0.f;
    for (int i = 0; i < D_ / 4; i++) {
        float4 a = xr[i], q = qr[i];
        acc += a.x * q.x + a.y * q.y + a.z * q.z + a.w * q.w;
    }
    float logit = acc * scale;
    red[l] = logit; __syncthreads();
    for (int off = 128; off > 0; off >>= 1) {
        if (l < off) red[l] = fmaxf(red[l], red[l + off]);
        __syncthreads();
    }
    float mx = red[0]; __syncthreads();
    float e = expf(logit - mx);
    attn[l] = e; red[l] = e; __syncthreads();
    for (int off = 128; off > 0; off >>= 1) {
        if (l < off) red[l] += red[l + off];
        __syncthreads();
    }
    float inv = 1.0f / red[0];

    const int d0 = l * 4;
    float4 a4 = {0.f, 0.f, 0.f, 0.f};
    for (int ll = 0; ll < L_; ll++) {
        float w = attn[ll] * inv;
        float4 xv = *(const float4*)(xbase + (size_t)ll * D_ + d0);
        a4.x += w * xv.x; a4.y += w * xv.y; a4.z += w * xv.z; a4.w += w * xv.w;
    }
    *(float4*)(xbar + (size_t)bs * D_ + d0) = a4;
}

// --- read attention + gate combine: yb = bf16(x + sig .* softmax(x.rk2^T).rv)
__global__ __launch_bounds__(256)
void k_read_pass(const float* __restrict__ x, const unsigned short* __restrict__ sigb,
                 const float* __restrict__ rk2, const float* __restrict__ rv,
                 unsigned short* __restrict__ yb) {
    const int tid = threadIdx.x;
    const int wid = tid >> 5, lane = tid & 31;   // wave per token, lane per slot
    const size_t bt = (size_t)blockIdx.x * 8 + wid;
    const int b  = (int)(bt >> 13);              // T_ = 8192
    const int tt = (int)(bt & (T_ - 1));
    const int seg = tt >> 8;                     // L_ = 256
    const float scale = 0.03125f;

    const float* xr  = x + bt * D_;
    const float* rkr = rk2 + ((size_t)b * S_ + lane) * D_;
    const bool act = (lane >= seg);
    float logit = -3.0e38f;
    {
        const float4* x4 = (const float4*)xr;
        const float4* k4 = (const float4*)rkr;
        float acc = 0.f;
        for (int i = 0; i < D_ / 4; i++) {
            float4 a = x4[i], q = k4[i];
            acc += a.x * q.x + a.y * q.y + a.z * q.z + a.w * q.w;
        }
        if (act) logit = acc * scale;
    }
    float m = logit;
    for (int off = 16; off > 0; off >>= 1) m = fmaxf(m, __shfl_xor(m, off, 32));
    float e = act ? expf(logit - m) : 0.f;
    float ssum = e;
    for (int off = 16; off > 0; off >>= 1) ssum += __shfl_xor(ssum, off, 32);
    float a = e / ssum;

    float4 acc8[8];
#pragma unroll
    for (int j = 0; j < 8; j++) { acc8[j].x = 0.f; acc8[j].y = 0.f; acc8[j].z = 0.f; acc8[j].w = 0.f; }
    for (int ss = 0; ss < S_; ss++) {
        float as = __shfl(a, ss, 32);
        const float4* rvr = (const float4*)(rv + ((size_t)b * S_ + ss) * D_ + lane * 32);
#pragma unroll
        for (int j = 0; j < 8; j++) {
            float4 vv = rvr[j];
            acc8[j].x += as * vv.x; acc8[j].y += as * vv.y;
            acc8[j].z += as * vv.z; acc8[j].w += as * vv.w;
        }
    }
    const float4*  xo = (const float4*)(xr + lane * 32);
    const ushort4* sg = (const ushort4*)(sigb + bt * D_ + lane * 32);
    ushort4*       yo = (ushort4*)(yb + bt * D_ + lane * 32);
#pragma unroll
    for (int j = 0; j < 8; j++) {
        float4 xv = xo[j];
        ushort4 sv = sg[j];
        float y0 = xv.x + bf2f(sv.x) * acc8[j].x;
        float y1 = xv.y + bf2f(sv.y) * acc8[j].y;
        float y2 = xv.z + bf2f(sv.z) * acc8[j].z;
        float y3 = xv.w + bf2f(sv.w) * acc8[j].w;
        ushort4 ov; ov.x = f2bf(y0); ov.y = f2bf(y1); ov.z = f2bf(y2); ov.w = f2bf(y3);
        yo[j] = ov;
    }
}

// ---------------------------------------------------------------- launcher
extern "C" void kernel_launch(void* const* d_in, const int* in_sizes, int n_in,
                              void* d_out, int out_size, void* d_ws, size_t ws_size,
                              hipStream_t stream) {
    const float* x            = (const float*)d_in[0];
    const float* gate_w       = (const float*)d_in[1];
    const float* gate_b       = (const float*)d_in[2];
    const float* read_q_w     = (const float*)d_in[3];
    const float* read_k_w     = (const float*)d_in[4];
    const float* read_v_w     = (const float*)d_in[5];
    const float* output_w     = (const float*)d_in[6];
    const float* memory_slots = (const float*)d_in[7];
    const float* write_q_w    = (const float*)d_in[8];
    const float* write_k_w    = (const float*)d_in[9];
    const float* write_v_w    = (const float*)d_in[10];

    float* out     = (float*)d_out;
    float* mem_out = out + (size_t)B_ * T_ * D_;     // second tuple output [B,S,D]

    char* ws = (char*)d_ws;
    const size_t nBTD = (size_t)B_ * T_ * D_;        // 33,554,432
    unsigned short* xb   = (unsigned short*)(ws + 0);            // 64 MB (reused as yb)
    unsigned short* sigb = (unsigned short*)(ws + (nBTD * 2));   // 64 MB
    unsigned short* wbg  = (unsigned short*)(ws + (nBTD * 4));   // 2 MB
    unsigned short* wbo  = (unsigned short*)(ws + (nBTD * 4) + 2097152);
    char* smallBase = ws + (nBTD * 4) + 4194304;
    float* qt   = (float*)(smallBase);                 // [32,1024]
    float* qk   = (float*)(smallBase + 131072);        // [32,1024]
    float* xbar = (float*)(smallBase + 262144);        // [128,1024]
    float* t1   = (float*)(smallBase + 262144 + 524288);
    float* rk2  = (float*)(smallBase + 262144 + 1048576);
    float* rv   = (float*)(smallBase + 262144 + 1572864);

    // 1) fp32 -> bf16 conversions
    k_f32_to_bf16<<<4096, 256, 0, stream>>>(x, xb, nBTD);
    k_f32_to_bf16<<<1024, 256, 0, stream>>>(gate_w, wbg, (size_t)D_ * D_);
    k_f32_to_bf16<<<1024, 256, 0, stream>>>(output_w, wbo, (size_t)D_ * D_);

    // 2) gate GEMM with fused sigmoid epilogue (bf16 WMMA, fp32 accum)
    dim3 gBig(D_ / 128, (B_ * T_) / 128);
    k_gemm_bf16<1><<<gBig, 256, 0, stream>>>(xb, wbg, gate_b, nullptr, sigb,
                                             B_ * T_, D_, D_);

    // 3-4) qk = (mem_slots @ Wq^T) @ Wk   (eliminates the write-k GEMM)
    k_sgemm_nt<<<(S_ * D_) / 256, 256, 0, stream>>>(memory_slots, write_q_w, qt, S_, D_, D_);
    k_sgemm_nn<<<(S_ * D_) / 256, 256, 0, stream>>>(qt, write_k_w, qk, S_, D_, D_);

    // 5) write attention -> xbar[b,s,:] = sum_l softmax(qk.x)_l * x_l
    k_write_pass<<<B_ * S_, 256, 0, stream>>>(x, qk, xbar);

    // 6) memory = xbar @ Wv^T  (exact fp32; eliminates the write-v GEMM)
    k_sgemm_nt<<<(B_ * S_ * D_) / 256, 256, 0, stream>>>(xbar, write_v_w, mem_out,
                                                         B_ * S_, D_, D_);

    // 7-9) rk2 = (memory @ Wk^T) @ Wq ; rv = memory @ Wv_read^T
    k_sgemm_nt<<<(B_ * S_ * D_) / 256, 256, 0, stream>>>(mem_out, read_k_w, t1, B_ * S_, D_, D_);
    k_sgemm_nn<<<(B_ * S_ * D_) / 256, 256, 0, stream>>>(t1, read_q_w, rk2, B_ * S_, D_, D_);
    k_sgemm_nt<<<(B_ * S_ * D_) / 256, 256, 0, stream>>>(mem_out, read_v_w, rv, B_ * S_, D_, D_);

    // 10) read attention + gate combine -> yb (bf16, reuses xb buffer)
    k_read_pass<<<(B_ * T_) / 8, 256, 0, stream>>>(x, sigb, rk2, rv, xb);

    // 11) out = y @ output_w^T (bf16 WMMA, fp32 result)
    k_gemm_bf16<0><<<gBig, 256, 0, stream>>>(xb, wbo, nullptr, out, nullptr,
                                             B_ * T_, D_, D_);
}
